// LaSAForNER_14310831030954
// MI455X (gfx1250) — matrix operand order repaired and checked by reading, not compiled, not run on hardware
//
#include <hip/hip_runtime.h>

// ---------------------------------------------------------------------------
// LaSA-for-NER forward (BERT-4L + chunk encoder + classifier + NLL loss)
// gfx1250 / CDNA5:
//   - GEMMs via v_wmma_f32_16x16x32_bf16 (wave32 WMMA, double-buffered LDS)
//   - chunk-attention staging via global_load_async_to_lds_b128 (ASYNCcnt)
//   - chunk scatter rows via TENSOR_LOAD_TO_LDS / TENSOR_STORE_FROM_LDS (TDM)
// ---------------------------------------------------------------------------

typedef __attribute__((ext_vector_type(16))) __bf16        v16bf;
typedef __attribute__((ext_vector_type(8)))  float         v8f;
typedef __attribute__((ext_vector_type(8)))  unsigned int  v8u;
typedef __attribute__((ext_vector_type(4)))  unsigned int  u32x4;
typedef __attribute__((ext_vector_type(8)))  int           i32x8;
typedef __attribute__((ext_vector_type(4)))  int           i32x4;

#define HDIM   768
#define NLAY   4
#define NHEAD  12
#define DFF    3072
#define BB     16
#define LL     256
#define NTOK   (BB*LL)          // 4096 rows
#define CN     64
#define CL     8
#define CNH    8
#define CDH    96               // 768/8
#define CDFF   2048
#define NLAB   9

// ---------------- helpers ----------------

__device__ __forceinline__ unsigned short f2bf(float f) {
    unsigned u = __builtin_bit_cast(unsigned, f);
    unsigned r = u + 0x7FFFu + ((u >> 16) & 1u);   // round-to-nearest-even
    return (unsigned short)(r >> 16);
}
__device__ __forceinline__ unsigned packbf(float lo, float hi) {
    return (unsigned)f2bf(lo) | ((unsigned)f2bf(hi) << 16);
}

__device__ __forceinline__ v8f wmma_bf16(v16bf a, v16bf b, v8f c) {
#if defined(__gfx1250__)
    return __builtin_amdgcn_wmma_f32_16x16x32_bf16(
        /*neg_a=*/false, a, /*neg_b=*/false, b,
        /*c_mod=*/(short)0, c, /*reuse_a=*/false, /*reuse_b=*/false);
#else
    (void)a; (void)b;
    return c;
#endif
}

// ---------------- WMMA GEMM ----------------
// C[M,N] = alpha * A[M,K] * op(B) + bias;  op(B)=B[K,N] or B^T with B[N,K].
// REQUIRES: K % 32 == 0, A (and B if TRANSB) rows 8-byte aligned.
// FULL: M % 128 == 0 && N % 64 == 0 -> all bounds guards compiled out.
// Strided-batched: z = zo*batch_inner + zi, base += zo*s*_o + zi*s*_i.
// Block: 256 threads (8 waves). Tile 128x64, K-step 32, double-buffered LDS.

#define BM 128
#define BN 64
#define BK 32
#define LPAD 20   // padded row stride (dwords): 80B keeps 16B alignment, spreads banks

template <bool TRANSB, bool FULL>
__device__ __forceinline__ void stage_tiles(
    unsigned (*__restrict__ As)[LPAD], unsigned (*__restrict__ Bs)[LPAD],
    const float* __restrict__ A, const float* __restrict__ Bm,
    int row0, int col0, int k0, int M, int N, int K,
    int lda, int ldb, int tid)
{
    // A tile: 128x32 fp32 -> bf16 pairs, As[m][kp]
    #pragma unroll
    for (int i = 0; i < 8; ++i) {
        int p   = tid + i * 256;          // 0..2047 pair slots
        int m   = p >> 4;
        int kp  = p & 15;
        int gm  = row0 + m;
        int gmc = FULL ? gm : (gm < M ? gm : M - 1);
        int gk  = k0 + kp * 2;
        const float2 f = *(const float2*)(A + (long long)gmc * lda + gk);
        unsigned pk = packbf(f.x, f.y);
        As[m][kp] = (FULL || gm < M) ? pk : 0u;
        if (i == 0 && k0 + BK < K)
            __builtin_prefetch(A + (long long)gmc * lda + k0 + BK);
    }
    // B tile: 32x64 -> Bs[n][kp] (per-lane-contiguous fragment rows)
    #pragma unroll
    for (int i = 0; i < 4; ++i) {
        int p   = tid + i * 256;          // 0..1023 pair slots
        int kp  = p >> 6;
        int n   = p & 63;
        int gk  = k0 + kp * 2;
        int gn  = col0 + n;
        int gnc = FULL ? gn : (gn < N ? gn : N - 1);
        float f0, f1;
        if (!TRANSB) {
            f0 = Bm[(long long)gk       * ldb + gnc];
            f1 = Bm[(long long)(gk + 1) * ldb + gnc];
        } else {
            const float2 f = *(const float2*)(Bm + (long long)gnc * ldb + gk);
            f0 = f.x; f1 = f.y;
        }
        unsigned pk = packbf(f0, f1);
        Bs[n][kp] = (FULL || gn < N) ? pk : 0u;
    }
}

template <bool TRANSB, bool FULL>
__global__ __launch_bounds__(256)
void k_gemm(const float* __restrict__ A, const float* __restrict__ Bm,
            const float* __restrict__ bias, float* __restrict__ C,
            int M, int N, int K, int batch_inner,
            long long sAo, long long sAi, long long sBo, long long sBi,
            long long sCo, long long sCi,
            int lda, int ldb, int ldc, float alpha)
{
    int z  = blockIdx.z;
    int zo = z / batch_inner, zi = z % batch_inner;
    A  += zo * sAo + zi * sAi;
    Bm += zo * sBo + zi * sBi;
    C  += zo * sCo + zi * sCi;

    __shared__ __align__(16) unsigned AsB[2][BM][LPAD];
    __shared__ __align__(16) unsigned BsB[2][BN][LPAD];

    const int tid  = threadIdx.x;
    const int lane = tid & 31;
    const int wave = tid >> 5;
    const int row0 = blockIdx.y * BM;
    const int col0 = blockIdx.x * BN;

    v8f acc[4] = {};

    const int mrow   = wave * 16 + (lane & 15);      // A-frag row in tile
    const int akp0   = (lane < 16) ? 0 : 8;          // A-frag K-pair base
    const int bkp0   = (lane < 16) ? 0 : 8;          // B-frag K-pair base
    const int ncol_l = lane & 15;                    // frag column

    // One MMA macro-step over the current LDS buffer: load A-frag + all 4
    // B-frags first (DS loads pipeline behind one wait), then 4 WMMAs.
    auto mma_step = [&](int cur) {
        const unsigned* arow = AsB[cur][mrow];
        uint4 a0 = *(const uint4*)(arow + akp0);
        uint4 a1 = *(const uint4*)(arow + akp0 + 4);
        uint4 b0[4], b1[4];
        #pragma unroll
        for (int j = 0; j < 4; ++j) {
            const unsigned* brow = BsB[cur][j * 16 + ncol_l];
            b0[j] = *(const uint4*)(brow + bkp0);
            b1[j] = *(const uint4*)(brow + bkp0 + 4);
        }
        v8u au;
        au[0] = a0.x; au[1] = a0.y; au[2] = a0.z; au[3] = a0.w;
        au[4] = a1.x; au[5] = a1.y; au[6] = a1.z; au[7] = a1.w;
        v16bf afrag = __builtin_bit_cast(v16bf, au);
        #pragma unroll
        for (int j = 0; j < 4; ++j) {
            v8u bu;
            bu[0] = b0[j].x; bu[1] = b0[j].y; bu[2] = b0[j].z; bu[3] = b0[j].w;
            bu[4] = b1[j].x; bu[5] = b1[j].y; bu[6] = b1[j].z; bu[7] = b1[j].w;
            acc[j] = wmma_bf16(afrag, __builtin_bit_cast(v16bf, bu), acc[j]);
        }
    };

    const int nsteps = K / BK;                       // K % 32 == 0 by contract

    stage_tiles<TRANSB, FULL>(AsB[0], BsB[0], A, Bm, row0, col0, 0, M, N, K, lda, ldb, tid);
    __syncthreads();

    int s = 0;
    for (; s < nsteps - 1; ++s) {                    // steady state: unconditional stage
        stage_tiles<TRANSB, FULL>(AsB[(s + 1) & 1], BsB[(s + 1) & 1], A, Bm,
                                  row0, col0, (s + 1) * BK, M, N, K, lda, ldb, tid);
        mma_step(s & 1);
        __syncthreads();
    }
    mma_step(s & 1);                                 // peeled last step

    // store per ISA 32-bit C/D 16x16 layout
    const int mbase = row0 + wave * 16 + ((lane < 16) ? 0 : 8);
    #pragma unroll
    for (int j = 0; j < 4; ++j) {
        int gc = col0 + j * 16 + ncol_l;
        if (FULL || gc < N) {
            float bv = bias ? bias[gc] : 0.f;
            #pragma unroll
            for (int r = 0; r < 8; ++r) {
                int gr = mbase + r;
                if (FULL || gr < M)
                    C[(long long)gr * ldc + gc] = acc[j][r] * alpha + bv;
            }
        }
    }
}

// ---------------- elementwise / reduction kernels ----------------

__global__ void k_zero(float* p, long long n) {
    long long i = (long long)blockIdx.x * blockDim.x + threadIdx.x;
    if (i < n) p[i] = 0.f;
}

__global__ __launch_bounds__(256)
void k_embed(const int* __restrict__ ids, const int* __restrict__ tt,
             const float* __restrict__ we, const float* __restrict__ pe,
             const float* __restrict__ te, float* __restrict__ out)
{
    int row = blockIdx.x;                 // 0..4095
    int l   = row & (LL - 1);
    int id  = ids[row];
    int t   = tt[row];
    const float* w = we + (long long)id * HDIM;
    const float* p = pe + (long long)l  * HDIM;
    const float* q = te + (long long)t  * HDIM;
    float* o = out + (long long)row * HDIM;
    for (int c = threadIdx.x; c < HDIM; c += blockDim.x)
        o[c] = w[c] + p[c] + q[c];
}

// out = LayerNorm(x + a) * g + b ; a may be null ; in-place safe (out==x)
__global__ __launch_bounds__(256)
void k_add_ln(const float* __restrict__ x, const float* __restrict__ a,
              const float* __restrict__ g, const float* __restrict__ b,
              float* __restrict__ out, int ncol, float eps)
{
    long long row = blockIdx.x;
    const float* xr = x + row * ncol;
    const float* ar = a ? (a + row * ncol) : nullptr;
    float vals[4];
    int   cnt  = 0;
    float lsum = 0.f;
    for (int c = threadIdx.x; c < ncol; c += blockDim.x) {
        float v = xr[c] + (ar ? ar[c] : 0.f);
        vals[cnt++] = v;
        lsum += v;
    }
    __shared__ float red[256];
    red[threadIdx.x] = lsum;
    __syncthreads();
    for (int st = 128; st > 0; st >>= 1) {
        if (threadIdx.x < st) red[threadIdx.x] += red[threadIdx.x + st];
        __syncthreads();
    }
    float mean = red[0] / ncol;
    __syncthreads();
    float lsq = 0.f;
    for (int i = 0; i < cnt; ++i) { float d = vals[i] - mean; lsq += d * d; }
    red[threadIdx.x] = lsq;
    __syncthreads();
    for (int st = 128; st > 0; st >>= 1) {
        if (threadIdx.x < st) red[threadIdx.x] += red[threadIdx.x + st];
        __syncthreads();
    }
    float rstd = rsqrtf(red[0] / ncol + eps);
    int idx = 0;
    for (int c = threadIdx.x; c < ncol; c += blockDim.x) {
        float v = (vals[idx++] - mean) * rstd;
        out[row * ncol + c] = v * g[c] + b[c];
    }
}

// softmax over last dim (==256==blockDim) with BERT mask bias
__global__ __launch_bounds__(256)
void k_softmax_mask(float* __restrict__ S, const int* __restrict__ mask,
                    int rows_per_b)
{
    int row = blockIdx.x;
    int b   = row / rows_per_b;
    float* sr = S + (long long)row * LL;
    int c = threadIdx.x;
    float biasv = (1.f - (float)mask[b * LL + c]) * -10000.f;
    float v = sr[c] + biasv;
    __shared__ float red[256];
    red[c] = v;
    __syncthreads();
    for (int st = 128; st > 0; st >>= 1) {
        if (c < st) red[c] = fmaxf(red[c], red[c + st]);
        __syncthreads();
    }
    float mx = red[0];
    __syncthreads();
    float e = expf(v - mx);
    red[c] = e;
    __syncthreads();
    for (int st = 128; st > 0; st >>= 1) {
        if (c < st) red[c] += red[c + st];
        __syncthreads();
    }
    sr[c] = e / red[0];
}

__global__ void k_gelu(float* p, long long n) {
    long long i = (long long)blockIdx.x * blockDim.x + threadIdx.x;
    if (i < n) {
        float x = p[i];
        float t = tanhf(0.7978845608f * (x + 0.044715f * x * x * x));
        p[i] = 0.5f * x * (1.f + t);
    }
}
__global__ void k_relu(float* p, long long n) {
    long long i = (long long)blockIdx.x * blockDim.x + threadIdx.x;
    if (i < n) p[i] = fmaxf(p[i], 0.f);
}

// ---------------- chunk encoder pieces ----------------

__global__ void k_chunk_offs(const int* __restrict__ clens, int* __restrict__ offs)
{
    int b = blockIdx.x, c = threadIdx.x;   // grid B, block CN
    int s = 0;
    for (int i = 0; i < c; ++i) s += clens[b * CN + i];
    offs[b * CN + c] = s;
}

__global__ __launch_bounds__(256)
void k_gather_feat(const float* __restrict__ tok, const int* __restrict__ clens,
                   const int* __restrict__ offs, float* __restrict__ feat,
                   int* __restrict__ validm)
{
    int r = blockIdx.x;                    // 0..4095 = ((b*CN)+c)*CL + l
    int b = r >> 9;
    int rem = r & 511;
    int c = rem >> 3;
    int l = rem & 7;
    int len   = clens[b * CN + c];
    int trunc = len < CL ? len : CL;
    int valid = (l < trunc) ? 1 : 0;
    int pos = offs[b * CN + c] + l;
    pos = pos < 0 ? 0 : (pos > LL - 1 ? LL - 1 : pos);
    const float* src = tok + ((long long)b * LL + pos) * HDIM;
    float* dst = feat + (long long)r * HDIM;
    for (int k = threadIdx.x; k < HDIM; k += blockDim.x)
        dst[k] = valid ? src[k] : 0.f;
    if (threadIdx.x == 0) validm[r] = valid;
}

// tiny 8x8 attention per (chunk, head); faithful mask bug: VALID keys -> -1e9.
// q/k/v tiles staged global->LDS with GLOBAL_LOAD_ASYNC_TO_LDS_B128 (ASYNCcnt).
__global__ __launch_bounds__(128)
void k_chunk_attn(const float* __restrict__ q, const float* __restrict__ k,
                  const float* __restrict__ v, const int* __restrict__ validm,
                  float* __restrict__ o)
{
    int z = blockIdx.x;                    // chunk*CNH + head
    int chunk = z >> 3;
    int h = z & 7;
    __shared__ __align__(16) float sh[3][CL][CDH];   // q,k,v: 3*8*96 fp32 = 9216B
    __shared__ float s[CL][CL], aw[CL][CL];
    int t = threadIdx.x;

#if defined(__gfx1250__)
    // 3 matrices x (8 rows x 24 16B-slots) = 576 async b128 copies; sh is laid
    // out so slot index maps linearly to the LDS destination.
    unsigned lds0 = (unsigned)(unsigned long long)&sh[0][0][0];
    #pragma unroll
    for (int i = 0; i < 5; ++i) {
        int slot = t + i * 128;
        if (slot < 576) {
            int mat = slot / 192;
            int r   = slot - mat * 192;
            int l   = r / 24;
            int c16 = r - l * 24;
            const float* bp = (mat == 0) ? q : ((mat == 1) ? k : v);
            const float* gp = bp + ((long long)chunk * CL + l) * HDIM + h * CDH + c16 * 4;
            unsigned long long ga = (unsigned long long)gp;
            unsigned dst = lds0 + (unsigned)slot * 16u;
            asm volatile("global_load_async_to_lds_b128 %0, %1, off"
                         :: "v"(dst), "v"(ga) : "memory");
        }
    }
    asm volatile("s_wait_asynccnt 0x0" ::: "memory");
#else
    for (int i = t; i < 3 * CL * CDH; i += 128) {
        int mat = i / (CL * CDH);
        int r   = i - mat * (CL * CDH);
        int l   = r / CDH, d = r % CDH;
        const float* bp = (mat == 0) ? q : ((mat == 1) ? k : v);
        sh[mat][l][d] = bp[((long long)chunk * CL + l) * HDIM + h * CDH + d];
    }
#endif
    __syncthreads();

    if (t < 64) {
        int qi = t >> 3, ki = t & 7;
        float acc = 0.f;
        for (int d = 0; d < CDH; ++d) acc += sh[0][qi][d] * sh[1][ki][d];
        float biasv = validm[chunk * CL + ki] ? -1e9f : 0.f;
        s[qi][ki] = acc * 0.1020620726f + biasv;   // 1/sqrt(96)
    }
    __syncthreads();
    if (t < 8) {
        float mx = s[t][0];
        for (int i = 1; i < 8; ++i) mx = fmaxf(mx, s[t][i]);
        float e[8], sm = 0.f;
        for (int i = 0; i < 8; ++i) { e[i] = expf(s[t][i] - mx); sm += e[i]; }
        for (int i = 0; i < 8; ++i) aw[t][i] = e[i] / sm;
    }
    __syncthreads();
    for (int i = t; i < CL * CDH; i += 128) {
        int qi = i / CDH, d = i % CDH;
        float acc = 0.f;
        for (int kk = 0; kk < 8; ++kk) acc += aw[qi][kk] * sh[2][kk][d];
        o[((long long)chunk * CL + qi) * HDIM + h * CDH + d] = acc;
    }
}

// Scatter encoded chunk rows back to the token grid via the Tensor Data Mover:
// one wave per row; contiguous 768-fp32 row bounced global->LDS->global with
// TENSOR_LOAD_TO_LDS / TENSOR_STORE_FROM_LDS, tracked by TENSORcnt.
__global__ __launch_bounds__(32)
void k_scatter_chunks(const float* __restrict__ xenc, const int* __restrict__ offs,
                      const int* __restrict__ validm, float* __restrict__ lasa)
{
    int r = blockIdx.x;                    // 0..4095
    __shared__ __align__(16) float tmp[HDIM];
    if (!validm[r]) return;
    int b = r >> 9;
    int rem = r & 511;
    int c = rem >> 3;
    int l = rem & 7;
    int pos = offs[b * CN + c] + l;
    if (pos < 0 || pos >= LL) return;      // mode='drop'
    const float* src = xenc + (long long)r * HDIM;
    float* dst = lasa + ((long long)b * LL + pos) * HDIM;

#if defined(__gfx1250__)
    unsigned ldsoff = (unsigned)(unsigned long long)&tmp[0];
    unsigned long long sa = (unsigned long long)src;
    unsigned long long da = (unsigned long long)dst;
    // D# group0: count=1 | lds_addr | global_addr[56:0] | type=2
    u32x4 gl0 = { 1u, ldsoff, (unsigned)sa,
                  (unsigned)((sa >> 32) & 0x01FFFFFFull) | (2u << 30) };
    u32x4 gs0 = { 1u, ldsoff, (unsigned)da,
                  (unsigned)((da >> 32) & 0x01FFFFFFull) | (2u << 30) };
    // D# group1: data_size=2 (4B); tensor_dim0=768; tensor_dim1=1;
    //            tile_dim0=768; tile_dim1=1; tensor_dim0_stride=768
    i32x8 g1 = { 0x20000, (int)(HDIM << 16), 1 << 16, (int)(HDIM << 16),
                 1, HDIM, 0, 0 };
    i32x4 gz4 = { 0, 0, 0, 0 };
#if __clang_major__ >= 23
    i32x8 gz8 = { 0, 0, 0, 0, 0, 0, 0, 0 };
    __builtin_amdgcn_tensor_load_to_lds(gl0, g1, gz4, gz4, gz8, 0);
    __builtin_amdgcn_s_wait_tensorcnt(0);
    __builtin_amdgcn_tensor_store_from_lds(gs0, g1, gz4, gz4, gz8, 0);
    __builtin_amdgcn_s_wait_tensorcnt(0);
#else
    __builtin_amdgcn_tensor_load_to_lds(gl0, g1, gz4, gz4, 0);
    __builtin_amdgcn_s_wait_tensorcnt(0);
    __builtin_amdgcn_tensor_store_from_lds(gs0, g1, gz4, gz4, 0);
    __builtin_amdgcn_s_wait_tensorcnt(0);
#endif
#else
    for (int k = threadIdx.x; k < HDIM; k += blockDim.x) dst[k] = src[k];
#endif
}

__global__ void k_valid_pos(const int* __restrict__ valid_ids, int* __restrict__ safe)
{
    int b = blockIdx.x, l = threadIdx.x;   // grid B, block L
    int s = 0;
    for (int i = 0; i <= l; ++i) s += valid_ids[b * LL + i];
    safe[b * LL + l] = (valid_ids[b * LL + l] == 1) ? (s - 1) : LL;
}

__global__ __launch_bounds__(256)
void k_compact(const float* __restrict__ lasa, const int* __restrict__ safe,
               float* __restrict__ out)
{
    int r = blockIdx.x;                    // b*L + l
    int b = r >> 8;
    int d = safe[r];
    if (d < 0 || d >= LL) return;
    const float* src = lasa + (long long)r * HDIM;
    float* dst = out + ((long long)b * LL + d) * HDIM;
    for (int k = threadIdx.x; k < HDIM; k += blockDim.x) dst[k] = src[k];
}

__global__ __launch_bounds__(256)
void k_loss_partial(const float* __restrict__ logits, const int* __restrict__ labels,
                    const int* __restrict__ lmask, float* __restrict__ accum)
{
    __shared__ float s_sum[256], s_cnt[256];
    int tid = threadIdx.x;
    float lsum = 0.f, lcnt = 0.f;
    for (int r = blockIdx.x * blockDim.x + tid; r < NTOK; r += gridDim.x * blockDim.x) {
        const float* lg = logits + (long long)r * NLAB;
        float mx = lg[0];
        for (int i = 1; i < NLAB; ++i) mx = fmaxf(mx, lg[i]);
        float sm = 0.f;
        for (int i = 0; i < NLAB; ++i) sm += expf(lg[i] - mx);
        int lab = labels[r];
        float logp = (lg[lab] - mx) - logf(sm);
        if (lmask[r] == 1 && lab != 0) { lsum += -logp; lcnt += 1.f; }
    }
    s_sum[tid] = lsum; s_cnt[tid] = lcnt;
    __syncthreads();
    for (int st = 128; st > 0; st >>= 1) {
        if (tid < st) { s_sum[tid] += s_sum[tid + st]; s_cnt[tid] += s_cnt[tid + st]; }
        __syncthreads();
    }
    if (tid == 0) { atomicAdd(&accum[0], s_sum[0]); atomicAdd(&accum[1], s_cnt[0]); }
}

__global__ void k_loss_final(const float* __restrict__ accum, float* __restrict__ out)
{
    if (threadIdx.x == 0) out[0] = accum[0] / fmaxf(accum[1], 1.f);
}

// ---------------- host-side orchestration ----------------

static inline void gemm(hipStream_t s, bool transb,
                        const float* A, const float* B, const float* bias, float* C,
                        int M, int N, int K, int batch, int binner,
                        long long sAo, long long sAi, long long sBo, long long sBi,
                        long long sCo, long long sCi,
                        int lda, int ldb, int ldc, float alpha)
{
    dim3 g((N + BN - 1) / BN, (M + BM - 1) / BM, batch);
    bool full = (M % BM == 0) && (N % BN == 0);
    if (full) {
        if (transb)
            k_gemm<true, true><<<g, 256, 0, s>>>(A, B, bias, C, M, N, K, binner,
                                                 sAo, sAi, sBo, sBi, sCo, sCi,
                                                 lda, ldb, ldc, alpha);
        else
            k_gemm<false, true><<<g, 256, 0, s>>>(A, B, bias, C, M, N, K, binner,
                                                  sAo, sAi, sBo, sBi, sCo, sCi,
                                                  lda, ldb, ldc, alpha);
    } else {
        if (transb)
            k_gemm<true, false><<<g, 256, 0, s>>>(A, B, bias, C, M, N, K, binner,
                                                  sAo, sAi, sBo, sBi, sCo, sCi,
                                                  lda, ldb, ldc, alpha);
        else
            k_gemm<false, false><<<g, 256, 0, s>>>(A, B, bias, C, M, N, K, binner,
                                                   sAo, sAi, sBo, sBi, sCo, sCi,
                                                   lda, ldb, ldc, alpha);
    }
}

static inline void gemm0(hipStream_t s, const float* A, const float* B,
                         const float* bias, float* C, int M, int N, int K)
{
    gemm(s, false, A, B, bias, C, M, N, K, 1, 1, 0, 0, 0, 0, 0, 0, K, N, N, 1.f);
}

static inline void zero(hipStream_t s, float* p, long long n)
{
    k_zero<<<(unsigned)((n + 255) / 256), 256, 0, s>>>(p, n);
}

extern "C" void kernel_launch(void* const* d_in, const int* in_sizes, int n_in,
                              void* d_out, int out_size, void* d_ws, size_t ws_size,
                              hipStream_t stream)
{
    (void)in_sizes; (void)n_in; (void)out_size; (void)ws_size;

    // ---- inputs (setup_inputs dict order; params flattened in insertion order)
    const int* ids    = (const int*)d_in[0];
    const int* amask  = (const int*)d_in[1];
    const int* ttids  = (const int*)d_in[2];
    const int* clens  = (const int*)d_in[3];
    const int* labels = (const int*)d_in[4];
    const int* lmask  = (const int*)d_in[5];
    const int* vids   = (const int*)d_in[6];
    const float* we   = (const float*)d_in[7];
    const float* pe   = (const float*)d_in[8];
    const float* te   = (const float*)d_in[9];
    const float* elg  = (const float*)d_in[10];
    const float* elb  = (const float*)d_in[11];
    // layers: base 12, 16 tensors each: wq,bq,wk,bk,wv,bv,wo,bo,ln1g,ln1b,w1,b1,w2,b2,ln2g,ln2b
    #define LPAR(i, k) ((const float*)d_in[12 + (i) * 16 + (k)])
    // 76 pool_w, 77 pool_b (pooled output unused), 78..93 enc, 94 cls_w, 95 cls_b
    #define EPAR(k)    ((const float*)d_in[78 + (k)])
    const float* cls_w = (const float*)d_in[94];
    const float* cls_b = (const float*)d_in[95];

    // ---- workspace layout (fp32) ----
    float* W = (float*)d_ws;
    const long long F    = (long long)NTOK * HDIM;        // 3,145,728
    const long long BIGN = (long long)NTOK * DFF;         // 12,582,912 (== 192*256*256)
    float* x    = W;            // activations / BERT token output
    float* fx   = W + 1 * F;    // chunk-encoder activations
    float* qb   = W + 2 * F;
    float* kb   = W + 3 * F;
    float* vb   = W + 4 * F;
    float* buf1 = W + 5 * F;
    float* buf2 = W + 6 * F;
    float* big  = W + 7 * F;    // attention scores / FFN hidden
    float* accum = W + 7 * F + BIGN;
    int* ioffs  = (int*)(accum + 8);
    int* validm = ioffs + BB * CN;
    int* safe   = validm + NTOK;

    float* out_loss   = (float*)d_out;
    float* out_logits = out_loss + 1;

    // ---- embeddings + LN ----
    k_embed<<<NTOK, 256, 0, stream>>>(ids, ttids, we, pe, te, x);
    k_add_ln<<<NTOK, 256, 0, stream>>>(x, nullptr, elg, elb, x, HDIM, 1e-12f);

    // ---- BERT layers ----
    const long long sTok = (long long)LL * HDIM;   // per-batch token stride
    for (int i = 0; i < NLAY; ++i) {
        gemm0(stream, x, LPAR(i, 0), LPAR(i, 1), qb, NTOK, HDIM, HDIM);
        gemm0(stream, x, LPAR(i, 2), LPAR(i, 3), kb, NTOK, HDIM, HDIM);
        gemm0(stream, x, LPAR(i, 4), LPAR(i, 5), vb, NTOK, HDIM, HDIM);
        // scores = Q K^T / 8   [B*NH, 256, 256] into `big`
        gemm(stream, true, qb, kb, nullptr, big, LL, LL, 64,
             BB * NHEAD, NHEAD,
             sTok, 64, sTok, 64,
             (long long)NHEAD * LL * LL, (long long)LL * LL,
             HDIM, HDIM, LL, 0.125f);
        k_softmax_mask<<<BB * NHEAD * LL, 256, 0, stream>>>(big, amask, NHEAD * LL);
        // attn = S V -> interleaved [B, L, NH*dh] in buf1
        gemm(stream, false, big, vb, nullptr, buf1, LL, 64, LL,
             BB * NHEAD, NHEAD,
             (long long)NHEAD * LL * LL, (long long)LL * LL,
             sTok, 64, sTok, 64,
             LL, HDIM, HDIM, 1.f);
        gemm0(stream, buf1, LPAR(i, 6), LPAR(i, 7), buf2, NTOK, HDIM, HDIM);
        k_add_ln<<<NTOK, 256, 0, stream>>>(x, buf2, LPAR(i, 8), LPAR(i, 9), x, HDIM, 1e-12f);
        gemm0(stream, x, LPAR(i, 10), LPAR(i, 11), big, NTOK, DFF, HDIM);
        k_gelu<<<(unsigned)((BIGN + 255) / 256), 256, 0, stream>>>(big, BIGN);
        gemm0(stream, big, LPAR(i, 12), LPAR(i, 13), buf2, NTOK, HDIM, DFF);
        k_add_ln<<<NTOK, 256, 0, stream>>>(x, buf2, LPAR(i, 14), LPAR(i, 15), x, HDIM, 1e-12f);
    }
    // x now holds BERT token output `tok`

    // ---- chunk encoder ----
    k_chunk_offs<<<BB, CN, 0, stream>>>(clens, ioffs);
    k_gather_feat<<<NTOK, 256, 0, stream>>>(x, clens, ioffs, fx, validm);
    gemm0(stream, fx, EPAR(0), EPAR(1), qb, NTOK, HDIM, HDIM);
    gemm0(stream, fx, EPAR(2), EPAR(3), kb, NTOK, HDIM, HDIM);
    gemm0(stream, fx, EPAR(4), EPAR(5), vb, NTOK, HDIM, HDIM);
    k_chunk_attn<<<BB * CN * CNH, 128, 0, stream>>>(qb, kb, vb, validm, buf1);
    gemm0(stream, buf1, EPAR(6), EPAR(7), buf2, NTOK, HDIM, HDIM);
    k_add_ln<<<NTOK, 256, 0, stream>>>(fx, buf2, EPAR(8), EPAR(9), fx, HDIM, 1e-5f);
    gemm0(stream, fx, EPAR(10), EPAR(11), big, NTOK, CDFF, HDIM);
    k_relu<<<(unsigned)(((long long)NTOK * CDFF + 255) / 256), 256, 0, stream>>>(big, (long long)NTOK * CDFF);
    gemm0(stream, big, EPAR(12), EPAR(13), buf2, NTOK, HDIM, CDFF);
    k_add_ln<<<NTOK, 256, 0, stream>>>(fx, buf2, EPAR(14), EPAR(15), fx, HDIM, 1e-5f);

    // scatter encoded chunks back to token grid (lasa in buf1) via TDM
    zero(stream, buf1, F);
    k_scatter_chunks<<<NTOK, 32, 0, stream>>>(fx, ioffs, validm, buf1);

    // ---- valid-token compaction (valid_out in buf2) ----
    k_valid_pos<<<BB, LL, 0, stream>>>(vids, safe);
    zero(stream, buf2, F);
    k_compact<<<NTOK, 256, 0, stream>>>(buf1, safe, buf2);

    // ---- classifier + loss ----
    gemm0(stream, buf2, cls_w, cls_b, out_logits, NTOK, NLAB, HDIM);
    zero(stream, accum, 8);
    k_loss_partial<<<BB, 256, 0, stream>>>(out_logits, labels, lmask, accum);
    k_loss_final<<<1, 32, 0, stream>>>(accum, out_loss);
}
